// Transformer_68573447848579
// MI455X (gfx1250) — matrix operand order repaired and checked by reading, not compiled, not run on hardware
//
#include <hip/hip_runtime.h>
#include <hip/hip_bf16.h>
#include <math.h>

// ---- model dims ----
#define TB   2
#define TS   1024
#define TD   1024
#define TL   4
#define TNH  16
#define TV   32000
#define THD  64
#define THID 2816
#define TBS  (TB * TS)

typedef __attribute__((ext_vector_type(16))) _Float16 v16h;
typedef __attribute__((ext_vector_type(8)))  _Float16 v8h;
typedef __attribute__((ext_vector_type(8)))  float    v8f;
typedef __attribute__((ext_vector_type(4)))  int      v4i;

// Async global->LDS path (gfx1250): guarded so the file compiles on any toolchain.
#if defined(__gfx1250__) && __has_builtin(__builtin_amdgcn_global_load_async_to_lds_b128) && __has_builtin(__builtin_amdgcn_s_wait_asynccnt)
#define USE_ASYNC_LDS 1
#else
#define USE_ASYNC_LDS 0
#endif

__device__ __forceinline__ v8f wmma_f16(v16h a, v16h b, v8f c) {
    return __builtin_amdgcn_wmma_f32_16x16x32_f16(false, a, false, b, (short)0, c, false, false);
}

#if USE_ASYNC_LDS
__device__ __forceinline__ void async_cp16(const _Float16* g, _Float16* l) {
    // builtin signature (probe-confirmed): (int4 addrspace(1)*, int4 addrspace(3)*, imm, imm)
    __builtin_amdgcn_global_load_async_to_lds_b128(
        (__attribute__((address_space(1))) v4i*)g,
        (__attribute__((address_space(3))) v4i*)l, 0, 0);
}
#endif

// ---------------- fragment helpers (ISA 7.12.2 wave32 layouts) ----------------
__device__ __forceinline__ v16h make_a(const v8h lo, const v8h hi) {
    v16h a;
#pragma unroll
    for (int i = 0; i < 8; ++i) { a[i] = lo[i]; a[8 + i] = hi[i]; }
    return a;
}

// A fragment from f32 source (used for probs in PV)
__device__ __forceinline__ v16h load_a_frag_f32(const float* __restrict__ A, int lda,
                                                int mbase, int kb, int lane) {
    const int half = lane >> 4;
    const int m    = mbase + (lane & 15);
    const float* p0 = A + (size_t)m * lda + kb + half * 8;
    const float* p1 = A + (size_t)m * lda + kb + 16 + half * 8;
    float buf[16];
    *(float4*)(buf + 0)  = ((const float4*)p0)[0];
    *(float4*)(buf + 4)  = ((const float4*)p0)[1];
    *(float4*)(buf + 8)  = ((const float4*)p1)[0];
    *(float4*)(buf + 12) = ((const float4*)p1)[1];
    v16h a;
#pragma unroll
    for (int i = 0; i < 16; ++i) a[i] = (_Float16)buf[i];
    return a;
}

// A fragment from f16 source, row stride lda
__device__ __forceinline__ v16h load_a_frag_h(const _Float16* __restrict__ A, int lda,
                                              int mbase, int kb, int lane) {
    const int half = lane >> 4;
    const int m    = mbase + (lane & 15);
    const _Float16* p0 = A + (size_t)m * lda + kb + half * 8;
    return make_a(*(const v8h*)p0, *(const v8h*)(p0 + 16));
}

// B (32x16 f16) NT: lane l<16 -> N=l, K 0..7,16..23; lane>=16 -> N=l-16, K 8..15,24..31
__device__ __forceinline__ v16h load_b_frag_nt_h(const _Float16* __restrict__ W, int ldw,
                                                 int nbase, int kb, int lane) {
    const int half = lane >> 4;
    const int n    = nbase + (lane & 15);
    const _Float16* p = W + (size_t)n * ldw + kb + half * 8;
    v16h b;
    v8h lo = *(const v8h*)p, hi = *(const v8h*)(p + 16);
#pragma unroll
    for (int i = 0; i < 8; ++i) { b[i] = lo[i]; b[8 + i] = hi[i]; }
    return b;
}

// C/D (16x16 f32): vgpr r, lane<16 -> M=r, N=lane ; lane>=16 -> M=r+8, N=lane-16
__device__ __forceinline__ void store_c_f32(float* __restrict__ C, int M, int ldc,
                                            int mbase, int nbase, int lane, v8f c) {
    const int half = lane >> 4;
    const int n    = nbase + (lane & 15);
#pragma unroll
    for (int r = 0; r < 8; ++r) {
        int m = mbase + r + half * 8;
        if (m < M) C[(size_t)m * ldc + n] = c[r];
    }
}
__device__ __forceinline__ void store_c_f16(_Float16* __restrict__ C, int M, int ldc,
                                            int mbase, int nbase, int lane, v8f c) {
    const int half = lane >> 4;
    const int n    = nbase + (lane & 15);
#pragma unroll
    for (int r = 0; r < 8; ++r) {
        int m = mbase + r + half * 8;
        if (m < M) C[(size_t)m * ldc + n] = (_Float16)c[r];
    }
}

// ---------------- tiled f16 NT GEMM: C[M,N] = A[M,K] * W[N,K]^T ----------------
// 256 threads = 8 waves (2 M x 4 N). Block tile 128x128, BK=32, double-buffered LDS.
// LDS rows padded to 40 halves -> conflict-free ds_load_b128 fragment reads.
#define BM  128
#define BN  128
#define BK  32
#define BKP 40

__global__ __launch_bounds__(256) void gemm_f16_nt(const _Float16* __restrict__ A,
                                                   const _Float16* __restrict__ W,
                                                   float* __restrict__ Cf,
                                                   _Float16* __restrict__ Ch,
                                                   int M, int N, int K,
                                                   int lda, int ldw, int ldc) {
    __shared__ _Float16 lsA[2][BM * BKP];   // 2 x 10 KB
    __shared__ _Float16 lsB[2][BN * BKP];   // 2 x 10 KB
    const int t    = threadIdx.x;
    const int lane = t & 31, wave = t >> 5;
    const int mw = wave >> 2, nw = wave & 3;
    const int mbase0 = blockIdx.y * BM, nbase0 = blockIdx.x * BN;
    const int half = lane >> 4, l15 = lane & 15;

    // per-thread copy descriptors: 2 chunks of 8 halves per tile per matrix
    int rr[2], ko[2];
    const _Float16 *gA[2], *gB[2];
#pragma unroll
    for (int c0 = 0; c0 < 2; ++c0) {
        int c = t + c0 * 256;
        rr[c0] = c >> 2;
        ko[c0] = (c & 3) * 8;
        int gm = mbase0 + rr[c0]; gm = gm < M ? gm : M - 1;
        int gn = nbase0 + rr[c0]; gn = gn < N ? gn : N - 1;
        gA[c0] = A + (size_t)gm * lda + ko[c0];
        gB[c0] = W + (size_t)gn * ldw + ko[c0];
    }

    auto issue = [&](int kb, int p) {
#pragma unroll
        for (int c0 = 0; c0 < 2; ++c0) {
#if USE_ASYNC_LDS
            async_cp16(gA[c0] + kb, &lsA[p][rr[c0] * BKP + ko[c0]]);
            async_cp16(gB[c0] + kb, &lsB[p][rr[c0] * BKP + ko[c0]]);
#else
            *(v8h*)(&lsA[p][rr[c0] * BKP + ko[c0]]) = *(const v8h*)(gA[c0] + kb);
            *(v8h*)(&lsB[p][rr[c0] * BKP + ko[c0]]) = *(const v8h*)(gB[c0] + kb);
#endif
        }
    };

    v8f acc[4][2] = {};
    const int nk = K / BK;
    issue(0, 0);

    for (int i = 0; i < nk; ++i) {
        const int cur = i & 1;
        if (i + 1 < nk) {
            issue((i + 1) * BK, cur ^ 1);
#if USE_ASYNC_LDS
            __builtin_amdgcn_s_wait_asynccnt(4);   // tile i landed; tile i+1 in flight
#endif
        } else {
#if USE_ASYNC_LDS
            __builtin_amdgcn_s_wait_asynccnt(0);
#endif
        }
        __syncthreads();   // all waves' tile-i data visible

        v16h bfr[2];
#pragma unroll
        for (int nt = 0; nt < 2; ++nt) {
            int n = nw * 32 + nt * 16 + l15;
            const _Float16* p = &lsB[cur][n * BKP + half * 8];
            bfr[nt] = make_a(*(const v8h*)p, *(const v8h*)(p + 16));
        }
#pragma unroll
        for (int mt = 0; mt < 4; ++mt) {
            int m = mw * 64 + mt * 16 + l15;
            const _Float16* p0 = &lsA[cur][m * BKP + half * 8];
            const _Float16* p1 = &lsA[cur][m * BKP + 16 + half * 8];
            v16h afr = make_a(*(const v8h*)p0, *(const v8h*)p1);
            acc[mt][0] = wmma_f16(afr, bfr[0], acc[mt][0]);
            acc[mt][1] = wmma_f16(afr, bfr[1], acc[mt][1]);
        }
        __syncthreads();   // reads done before buffer 'cur' is overwritten at i+2
    }

#pragma unroll
    for (int mt = 0; mt < 4; ++mt)
#pragma unroll
        for (int nt = 0; nt < 2; ++nt) {
            int mb = mbase0 + mw * 64 + mt * 16;
            int nb = nbase0 + nw * 32 + nt * 16;
            if (Ch) store_c_f16(Ch, M, ldc, mb, nb, lane, acc[mt][nt]);
            else    store_c_f32(Cf, M, ldc, mb, nb, lane, acc[mt][nt]);
        }
}

// ---------------- attention (per-b launch, grid.z = head) ----------------
// scores[h,i,j] = sum_d q[i,h,d]*k[j,h,d]   (q,k f16 slices of one batch)
__global__ void attn_qk_kernel(const _Float16* __restrict__ q, const _Float16* __restrict__ k,
                               float* __restrict__ scores) {
    const int h = blockIdx.z;
    const _Float16* A = q + h * THD;
    const _Float16* W = k + h * THD;
    float* C = scores + (size_t)h * TS * TS;
    const int lane  = threadIdx.x & 31;
    const int wave  = threadIdx.x >> 5;
    const int nbase = (blockIdx.x * 8 + wave) * 16;
    const int mbase = blockIdx.y * 16;
    v8f acc = {};
#pragma unroll
    for (int kb = 0; kb < THD; kb += 32) {
        v16h a  = load_a_frag_h(A, TD, mbase, kb, lane);
        v16h bf = load_b_frag_nt_h(W, TD, nbase, kb, lane);
        acc = wmma_f16(a, bf, acc);
    }
    store_c_f32(C, TS, TS, mbase, nbase, lane, acc);
}

// o[i,h,d] = sum_j P[h,i,j] * vt[h,d,j]  (P f32 probs, vt f16 pre-transposed, o f16)
__global__ void attn_pv_kernel(const float* __restrict__ p, const _Float16* __restrict__ vt,
                               _Float16* __restrict__ o) {
    const int h = blockIdx.z;
    const float* A = p + (size_t)h * TS * TS;
    const _Float16* Vt = vt + (size_t)h * THD * TS;   // [HD, S] row-major
    _Float16* C = o + h * THD;
    const int lane  = threadIdx.x & 31;
    const int wave  = threadIdx.x >> 5;
    const int nbase = (blockIdx.x * 8 + wave) * 16;
    const int mbase = blockIdx.y * 16;
    if (nbase >= THD) return;
    v8f acc = {};
    for (int kb = 0; kb < TS; kb += 32) {
        v16h a  = load_a_frag_f32(A, TS, mbase, kb, lane);
        v16h bf = load_b_frag_nt_h(Vt, TS, nbase, kb, lane);
        acc = wmma_f16(a, bf, acc);
    }
    store_c_f16(C, TS, TD, mbase, nbase, lane, acc);
}

// v[B,S,NH,HD] -> vt[B,NH,HD,S]
__global__ void transpose_v_kernel(const _Float16* __restrict__ v, _Float16* __restrict__ vt) {
    int idx = blockIdx.x * 256 + threadIdx.x;
    if (idx >= TBS * TD) return;
    int d = idx % THD;
    int r = idx / THD;
    int hh = r % TNH; r /= TNH;
    int s = r % TS;
    int b = r / TS;
    vt[(((size_t)b * TNH + hh) * THD + d) * TS + s] = v[idx];
}

// ---------------- elementwise / norm / softmax ----------------
__global__ void cvt_f32_f16_kernel(const float* __restrict__ src, _Float16* __restrict__ dst,
                                   int n) {
    int idx = blockIdx.x * 256 + threadIdx.x;
    if (idx < n) dst[idx] = (_Float16)src[idx];
}

__global__ void embed_kernel(const int* __restrict__ tokens, const float* __restrict__ emb,
                             float* __restrict__ h) {
    int idx = blockIdx.x * 256 + threadIdx.x;
    if (idx >= TBS * TD) return;
    int t = idx / TD, d = idx % TD;
    h[idx] = emb[(size_t)tokens[t] * TD + d];
}

__global__ void rmsnorm_kernel(const float* __restrict__ x, long xstride,
                               const float* __restrict__ w, _Float16* __restrict__ y, int dim) {
    const int row = blockIdx.x;
    const float* xr = x + (size_t)row * xstride;
    __shared__ float red[256];
    float s = 0.f;
    for (int i = threadIdx.x; i < dim; i += 256) { float t = xr[i]; s += t * t; }
    red[threadIdx.x] = s;
    __syncthreads();
    for (int off = 128; off > 0; off >>= 1) {
        if (threadIdx.x < off) red[threadIdx.x] += red[threadIdx.x + off];
        __syncthreads();
    }
    float inv = rsqrtf(red[0] / dim + 1e-5f);
    for (int i = threadIdx.x; i < dim; i += 256)
        y[(size_t)row * dim + i] = (_Float16)(xr[i] * inv * w[i]);
}

__global__ void rope_kernel(_Float16* __restrict__ x) {
    // x: [B,S,NH,HD] f16; one thread per rotation pair
    int idx = blockIdx.x * 256 + threadIdx.x;
    const int total = TBS * TNH * (THD / 2);
    if (idx >= total) return;
    int f = idx % (THD / 2);
    int r = idx / (THD / 2);
    r /= TNH;
    int s = r % TS;
    float freq = powf(10000.0f, -(float)(2 * f) / (float)THD);
    float fr = (float)s * freq;
    float c = cosf(fr), sn = sinf(fr);
    _Float16* p = x + (size_t)idx * 2;
    float a = (float)p[0], b = (float)p[1];
    p[0] = (_Float16)(a * c - b * sn);
    p[1] = (_Float16)(a * sn + b * c);
}

__global__ void softmax_kernel(float* __restrict__ scores) {
    // blockIdx.x = query row i, blockIdx.y = head
    const int i = blockIdx.x, z = blockIdx.y;
    float* row = scores + ((size_t)z * TS + i) * TS;
    const float NEG = -20.72326583694641f;   // logf(1e-9f)
    __shared__ float red[256];
    float vals[4];
    float mx = -1e30f;
#pragma unroll
    for (int t = 0; t < 4; ++t) {
        int j = t * 256 + threadIdx.x;
        float m = (j == 0 || j > i) ? NEG : 0.0f;
        float vv = row[j] * 0.125f + m;
        vals[t] = vv;
        mx = fmaxf(mx, vv);
    }
    red[threadIdx.x] = mx;
    __syncthreads();
    for (int off = 128; off > 0; off >>= 1) {
        if (threadIdx.x < off) red[threadIdx.x] = fmaxf(red[threadIdx.x], red[threadIdx.x + off]);
        __syncthreads();
    }
    mx = red[0];
    __syncthreads();
    float sum = 0.f;
#pragma unroll
    for (int t = 0; t < 4; ++t) { vals[t] = expf(vals[t] - mx); sum += vals[t]; }
    red[threadIdx.x] = sum;
    __syncthreads();
    for (int off = 128; off > 0; off >>= 1) {
        if (threadIdx.x < off) red[threadIdx.x] += red[threadIdx.x + off];
        __syncthreads();
    }
    float inv = 1.0f / red[0];
#pragma unroll
    for (int t = 0; t < 4; ++t) {
        int j = t * 256 + threadIdx.x;
        row[j] = vals[t] * inv;
    }
}

__global__ void swiglu_kernel(const float* __restrict__ a, const float* __restrict__ b,
                              _Float16* __restrict__ out, int n) {
    int idx = blockIdx.x * 256 + threadIdx.x;
    if (idx >= n) return;
    float x = a[idx];
    float s = x / (1.0f + expf(-x));
    out[idx] = (_Float16)(s * b[idx]);
}

__global__ void add_kernel(float* __restrict__ dst, const float* __restrict__ src, int n) {
    int idx = blockIdx.x * 256 + threadIdx.x;
    if (idx >= n) return;
    dst[idx] += src[idx];
}

// ---------------- host orchestration ----------------
static inline void launch_gemm_f32out(const _Float16* A, const _Float16* W, float* C,
                                      int M, int N, int K, int lda, int ldw, int ldc,
                                      hipStream_t s) {
    dim3 grid(N / BN, (M + BM - 1) / BM, 1);
    gemm_f16_nt<<<grid, 256, 0, s>>>(A, W, C, nullptr, M, N, K, lda, ldw, ldc);
}
static inline void launch_gemm_f16out(const _Float16* A, const _Float16* W, _Float16* C,
                                      int M, int N, int K, int lda, int ldw, int ldc,
                                      hipStream_t s) {
    dim3 grid(N / BN, (M + BM - 1) / BM, 1);
    gemm_f16_nt<<<grid, 256, 0, s>>>(A, W, nullptr, C, M, N, K, lda, ldw, ldc);
}

extern "C" void kernel_launch(void* const* d_in, const int* in_sizes, int n_in,
                              void* d_out, int out_size, void* d_ws, size_t ws_size,
                              hipStream_t stream) {
    const int*   tokens  = (const int*)d_in[0];
    const float* tok_emb = (const float*)d_in[2];
    const float* wq      = (const float*)d_in[3];
    const float* wk      = (const float*)d_in[4];
    const float* wv      = (const float*)d_in[5];
    const float* wo      = (const float*)d_in[6];
    const float* w1      = (const float*)d_in[7];
    const float* w2      = (const float*)d_in[8];
    const float* w3      = (const float*)d_in[9];
    const float* anw     = (const float*)d_in[10];
    const float* fnw     = (const float*)d_in[11];
    const float* finw    = (const float*)d_in[12];
    const float* outw    = (const float*)d_in[13];
    float* out = (float*)d_out;

    // ---- workspace carve-up: f32 region, then f16 region ----
    float* ws = (float*)d_ws;
    size_t o = 0;
    float* h    = ws + o; o += (size_t)TBS * TD;           // residual stream
    float* tmp  = ws + o; o += (size_t)TBS * TD;           // gemm f32 out
    float* ff1  = ws + o; o += (size_t)TBS * THID;
    float* ff3  = ws + o; o += (size_t)TBS * THID;
    float* sc   = ws + o; o += (size_t)TNH * TS * TS;      // per-b scores

    _Float16* hb = (_Float16*)(ws + o);
    size_t ho = 0;
    _Float16* xn_h   = hb + ho; ho += (size_t)TBS * TD;
    _Float16* q_h    = hb + ho; ho += (size_t)TBS * TD;
    _Float16* k_h    = hb + ho; ho += (size_t)TBS * TD;
    _Float16* v_h    = hb + ho; ho += (size_t)TBS * TD;
    _Float16* vt_h   = hb + ho; ho += (size_t)TBS * TD;    // [B,NH,HD,S]
    _Float16* ao_h   = hb + ho; ho += (size_t)TBS * TD;
    _Float16* ff1_h  = hb + ho; ho += (size_t)TBS * THID;
    _Float16* hlast  = hb + ho; ho += (size_t)TB * TD;
    _Float16* wq_h   = hb + ho; ho += (size_t)TL * TD * TD;
    _Float16* wk_h   = hb + ho; ho += (size_t)TL * TD * TD;
    _Float16* wv_h   = hb + ho; ho += (size_t)TL * TD * TD;
    _Float16* wo_h   = hb + ho; ho += (size_t)TL * TD * TD;
    _Float16* w1_h   = hb + ho; ho += (size_t)TL * THID * TD;
    _Float16* w2_h   = hb + ho; ho += (size_t)TL * TD * THID;
    _Float16* w3_h   = hb + ho; ho += (size_t)TL * THID * TD;
    _Float16* outw_h = hb + ho; ho += (size_t)TV * TD;

    // ---- one-shot weight conversion to f16 ----
    {
        struct { const float* s; _Float16* d; size_t n; } cv[] = {
            {wq, wq_h, (size_t)TL * TD * TD},   {wk, wk_h, (size_t)TL * TD * TD},
            {wv, wv_h, (size_t)TL * TD * TD},   {wo, wo_h, (size_t)TL * TD * TD},
            {w1, w1_h, (size_t)TL * THID * TD}, {w2, w2_h, (size_t)TL * TD * THID},
            {w3, w3_h, (size_t)TL * THID * TD}, {outw, outw_h, (size_t)TV * TD},
        };
        for (auto& c : cv)
            cvt_f32_f16_kernel<<<((int)c.n + 255) / 256, 256, 0, stream>>>(c.s, c.d, (int)c.n);
    }

    const int nBSD = TBS * TD;
    embed_kernel<<<(nBSD + 255) / 256, 256, 0, stream>>>(tokens, tok_emb, h);

    for (int l = 0; l < TL; ++l) {
        const _Float16* lwq = wq_h + (size_t)l * TD * TD;
        const _Float16* lwk = wk_h + (size_t)l * TD * TD;
        const _Float16* lwv = wv_h + (size_t)l * TD * TD;
        const _Float16* lwo = wo_h + (size_t)l * TD * TD;
        const _Float16* lw1 = w1_h + (size_t)l * THID * TD;
        const _Float16* lw2 = w2_h + (size_t)l * TD * THID;
        const _Float16* lw3 = w3_h + (size_t)l * THID * TD;

        // ---- attention ----
        rmsnorm_kernel<<<TBS, 256, 0, stream>>>(h, TD, anw + (size_t)l * TD, xn_h, TD);
        launch_gemm_f16out(xn_h, lwq, q_h, TBS, TD, TD, TD, TD, TD, stream);
        launch_gemm_f16out(xn_h, lwk, k_h, TBS, TD, TD, TD, TD, TD, stream);
        launch_gemm_f16out(xn_h, lwv, v_h, TBS, TD, TD, TD, TD, TD, stream);

        const int npairs = TBS * TNH * (THD / 2);
        rope_kernel<<<(npairs + 255) / 256, 256, 0, stream>>>(q_h);
        rope_kernel<<<(npairs + 255) / 256, 256, 0, stream>>>(k_h);
        transpose_v_kernel<<<(nBSD + 255) / 256, 256, 0, stream>>>(v_h, vt_h);

        for (int b = 0; b < TB; ++b) {
            const _Float16* qb  = q_h  + (size_t)b * TS * TD;
            const _Float16* kb  = k_h  + (size_t)b * TS * TD;
            const _Float16* vtb = vt_h + (size_t)b * TS * TD;   // b*NH*HD*S
            _Float16*       ob  = ao_h + (size_t)b * TS * TD;
            attn_qk_kernel<<<dim3(TS / 16 / 8, TS / 16, TNH), 256, 0, stream>>>(qb, kb, sc);
            softmax_kernel<<<dim3(TS, TNH), 256, 0, stream>>>(sc);
            attn_pv_kernel<<<dim3(1, TS / 16, TNH), 256, 0, stream>>>(sc, vtb, ob);
        }

        launch_gemm_f32out(ao_h, lwo, tmp, TBS, TD, TD, TD, TD, TD, stream);
        add_kernel<<<(nBSD + 255) / 256, 256, 0, stream>>>(h, tmp, nBSD);

        // ---- FFN ----
        rmsnorm_kernel<<<TBS, 256, 0, stream>>>(h, TD, fnw + (size_t)l * TD, xn_h, TD);
        launch_gemm_f32out(xn_h, lw1, ff1, TBS, THID, TD, TD, TD, THID, stream);
        launch_gemm_f32out(xn_h, lw3, ff3, TBS, THID, TD, TD, TD, THID, stream);
        const int nFF = TBS * THID;
        swiglu_kernel<<<(nFF + 255) / 256, 256, 0, stream>>>(ff1, ff3, ff1_h, nFF);
        launch_gemm_f32out(ff1_h, lw2, tmp, TBS, TD, THID, THID, THID, TD, stream);
        add_kernel<<<(nBSD + 255) / 256, 256, 0, stream>>>(h, tmp, nBSD);
    }

    // ---- final norm on last token, then logits ----
    rmsnorm_kernel<<<TB, 256, 0, stream>>>(h + (size_t)(TS - 1) * TD, (long)TS * TD,
                                           finw, hlast, TD);
    launch_gemm_f32out(hlast, outw_h, out, TB, TV, TD, TD, TD, TV, stream);
}
